// CrossAttention_57767310131763
// MI455X (gfx1250) — matrix operand order repaired and checked by reading, not compiled, not run on hardware
//
#include <hip/hip_runtime.h>

typedef __attribute__((ext_vector_type(16))) __bf16        v16bf;
typedef __attribute__((ext_vector_type(2)))  __bf16        v2bf;
typedef __attribute__((ext_vector_type(8)))  float         v8f;
typedef __attribute__((ext_vector_type(4)))  unsigned int  u32x4;
typedef __attribute__((ext_vector_type(4)))  float         f32x4;

#define BB  4
#define CC  256
#define HW  4096
#define CQK 32
#define KSPLIT 4

// WMMA operand fragment: 16 bf16 = 32 bytes = 8 VGPRs
union Frag {
  v16bf v;
  u32x4 q[2];
  unsigned u[8];
};

// pack two f32 -> two bf16 (RNE), packed in a dword
#if __has_builtin(__builtin_amdgcn_cvt_pk_bf16_f32)
__device__ __forceinline__ unsigned f2bf2(float lo, float hi) {
  v2bf p = __builtin_amdgcn_cvt_pk_bf16_f32(lo, hi);
  return __builtin_bit_cast(unsigned, p);
}
#else
__device__ __forceinline__ unsigned f2bf2(float lo, float hi) {
  unsigned a = __float_as_uint(lo), b = __float_as_uint(hi);
  a = (a + 0x7fffu + ((a >> 16) & 1u)) >> 16;
  b = (b + 0x7fffu + ((b >> 16) & 1u)) & 0xffff0000u;
  return (a & 0xffffu) | b;
}
#endif
__device__ __forceinline__ unsigned short f2bf(float x) {
  return (unsigned short)(f2bf2(x, 0.f) & 0xffffu);
}

// Load a 16-bit A-operand fragment (M=16 rows in lanes, K=32) from an f32
// row-major matrix, converting to bf16.
// Layout: lanes 0-15 hold K 0..15, lanes 16-31 hold K 16..31.
__device__ __forceinline__ Frag load_A_f32(const float* __restrict__ row, int lg) {
  Frag a;
  const float* p = row + lg * 16;
  f32x4 f0 = *(const f32x4*)(p);
  f32x4 f1 = *(const f32x4*)(p + 4);
  f32x4 f2 = *(const f32x4*)(p + 8);
  f32x4 f3 = *(const f32x4*)(p + 12);
  a.u[0] = f2bf2(f0.x, f0.y); a.u[1] = f2bf2(f0.z, f0.w);
  a.u[2] = f2bf2(f1.x, f1.y); a.u[3] = f2bf2(f1.z, f1.w);
  a.u[4] = f2bf2(f2.x, f2.y); a.u[5] = f2bf2(f2.z, f2.w);
  a.u[6] = f2bf2(f3.x, f3.y); a.u[7] = f2bf2(f3.z, f3.w);
  return a;
}

// ---------------------------------------------------------------------------
// q/k projection: qT,kT[b][n][o] (bf16, channel-fastest) = Wq/Wk @ depth + b
// ---------------------------------------------------------------------------
__global__ __launch_bounds__(128) void proj_qk_kernel(
    const float* __restrict__ depth,
    const float* __restrict__ Wq, const float* __restrict__ bq,
    const float* __restrict__ Wk, const float* __restrict__ bk,
    unsigned short* __restrict__ qT, unsigned short* __restrict__ kT) {
  const int b    = blockIdx.x / (HW / 64);
  const int nblk = blockIdx.x % (HW / 64);
  const int wave = threadIdx.x >> 5;
  const int lane = threadIdx.x & 31;
  const int l = lane & 15, lg = lane >> 4;
  const int n = nblk * 64 + wave * 16 + l;

  v8f acc[4] = {};  // qM0, qM1, kM0, kM1

  for (int k0 = 0; k0 < CC; k0 += 32) {
    Frag A[4];
#pragma unroll
    for (int t = 0; t < 4; ++t) {
      const float* W = (t < 2) ? Wq : Wk;
      A[t] = load_A_f32(W + ((t & 1) * 16 + l) * CC + k0, lg);
    }
    // B operand: depth tile (K=c, N=n); lanes 0-15: K=0..15, 16-31: K=16..31
    Frag Bf;
    const float* dp = depth + ((size_t)b * CC + k0 + lg * 16) * HW + n;
#pragma unroll
    for (int j = 0; j < 8; ++j)
      Bf.u[j] = f2bf2(dp[(2 * j) * HW], dp[(2 * j + 1) * HW]);
#pragma unroll
    for (int t = 0; t < 4; ++t)
      acc[t] = __builtin_amdgcn_wmma_f32_16x16x32_bf16(
          false, A[t].v, false, Bf.v, (short)0, acc[t], false, false);
  }
#pragma unroll
  for (int t = 0; t < 4; ++t) {
    const float* bias = (t < 2) ? bq : bk;
    unsigned short* out = (t < 2) ? qT : kT;
#pragma unroll
    for (int j = 0; j < 8; ++j) {
      int o = (t & 1) * 16 + lg * 8 + j;
      out[((size_t)b * HW + n) * CQK + o] = f2bf(acc[t][j] + bias[o]);
    }
  }
}

// ---------------------------------------------------------------------------
// v projection: v[b][c][n] (bf16, token-fastest) = Wv @ rgb + bv
// ---------------------------------------------------------------------------
__global__ __launch_bounds__(128) void proj_v_kernel(
    const float* __restrict__ rgb,
    const float* __restrict__ Wv, const float* __restrict__ bv,
    unsigned short* __restrict__ v) {
  const int nb = HW / 64, cb = CC / 64;
  int idx = blockIdx.x;
  const int b = idx / (nb * cb);
  idx %= nb * cb;
  const int nblk = idx / cb, cblk = idx % cb;
  const int wave = threadIdx.x >> 5;
  const int lane = threadIdx.x & 31;
  const int l = lane & 15, lg = lane >> 4;
  const int c0 = cblk * 64 + wave * 16;
  const int n0 = nblk * 64;

  v8f acc[4] = {};
  for (int k0 = 0; k0 < CC; k0 += 32) {
    Frag A = load_A_f32(Wv + (c0 + l) * CC + k0, lg);
    Frag Bf[4];
#pragma unroll
    for (int t = 0; t < 4; ++t) {
      const float* rp = rgb + ((size_t)b * CC + k0 + lg * 16) * HW + n0 + t * 16 + l;
#pragma unroll
      for (int j = 0; j < 8; ++j)
        Bf[t].u[j] = f2bf2(rp[(2 * j) * HW], rp[(2 * j + 1) * HW]);
    }
#pragma unroll
    for (int t = 0; t < 4; ++t)
      acc[t] = __builtin_amdgcn_wmma_f32_16x16x32_bf16(
          false, A.v, false, Bf[t].v, (short)0, acc[t], false, false);
  }
#pragma unroll
  for (int t = 0; t < 4; ++t)
#pragma unroll
    for (int j = 0; j < 8; ++j) {
      int c = c0 + lg * 8 + j;
      v[((size_t)b * CC + c) * HW + n0 + t * 16 + l] = f2bf(acc[t][j] + bv[c]);
    }
}

// ---------------------------------------------------------------------------
// Fused flash attention with 4-way split-K over keys.
// Block = 4 waves on one 16-query tile; wave w scans keys [w*1024, (w+1)*1024),
// online softmax, O accumulated transposed (M=c, N=n) in 128 VGPRs.
// Partials merged across waves in LDS (ds_add_f32), then normalized store.
// ---------------------------------------------------------------------------
__global__ __launch_bounds__(128) void attn_kernel(
    const unsigned short* __restrict__ qT, const unsigned short* __restrict__ kT,
    const unsigned short* __restrict__ v, float* __restrict__ out) {
  __shared__ __align__(16) unsigned short P[KSPLIT][16][72];  // per-wave P^T staging
  __shared__ float redM[KSPLIT][16];
  __shared__ float redL[KSPLIT][16];
  __shared__ float sInvL[16];
  __shared__ __align__(16) float sO[CC * 16];  // [c][n], 16 KB

  const int b    = blockIdx.x / (HW / 16);
  const int nt   = blockIdx.x % (HW / 16);
  const int wave = threadIdx.x >> 5;
  const int lane = threadIdx.x & 31;
  const int l = lane & 15, lg = lane >> 4;
  const int n0 = nt * 16;

  // persistent q B-fragment (K=c 0..31, N=n): lanes 0-15 K=0..15, 16-31 K=16..31
  Frag Q;
  {
    const unsigned short* qp = qT + ((size_t)b * HW + n0 + l) * CQK + lg * 16;
    Q.q[0] = *(const u32x4*)(qp);
    Q.q[1] = *(const u32x4*)(qp + 8);
  }

  v8f O[16] = {};
  float mrun = -1e30f, lrun = 0.f;

  const int mbase = wave * (HW / KSPLIT);
  for (int mi = 0; mi < HW / KSPLIT; mi += 64) {
    const int m0 = mbase + mi;
    // S^T tiles: St(M=m, N=n) = kT-rows x qT-cols, K=Cqk=32 in one WMMA each
    v8f st[4];
#pragma unroll
    for (int s = 0; s < 4; ++s) {
      Frag Kf;
      const unsigned short* kp = kT + ((size_t)b * HW + m0 + s * 16 + l) * CQK + lg * 16;
      Kf.q[0] = *(const u32x4*)(kp);
      Kf.q[1] = *(const u32x4*)(kp + 8);
      v8f z = {};
      st[s] = __builtin_amdgcn_wmma_f32_16x16x32_bf16(
          false, Kf.v, false, Q.v, (short)0, z, false, false);
    }
    // online softmax: n lives in the lane -> one xor-16 shuffle per reduction
    float lm = -1e30f;
#pragma unroll
    for (int s = 0; s < 4; ++s)
#pragma unroll
      for (int j = 0; j < 8; ++j) lm = fmaxf(lm, st[s][j]);
    lm = fmaxf(lm, __shfl_xor(lm, 16));
    // rescale only when some lane's max actually grew (wave-uniform branch)
    if (__ballot(lm > mrun) != 0ull) {
      float mnew  = fmaxf(mrun, lm);
      float alpha = __expf(mrun - mnew);
      mrun = mnew;
      lrun *= alpha;
#pragma unroll
      for (int ct = 0; ct < 16; ++ct)
#pragma unroll
        for (int j = 0; j < 8; ++j) O[ct][j] *= alpha;
    }
    float ls = 0.f;
#pragma unroll
    for (int s = 0; s < 4; ++s) {
      float e[8];
#pragma unroll
      for (int j = 0; j < 8; ++j) { e[j] = __expf(st[s][j] - mrun); ls += e[j]; }
#pragma unroll
      for (int t = 0; t < 4; ++t)
        *(unsigned*)&P[wave][l][s * 16 + lg * 8 + 2 * t] = f2bf2(e[2 * t], e[2 * t + 1]);
    }
    ls += __shfl_xor(ls, 16);
    lrun += ls;

    // wave-private LDS round trip; DS ops are issue-ordered within a wave
    asm volatile("" ::: "memory");

    // P^T B-fragments (K=m, N=n), one per 32-key half — shared by all 16 c-tiles
    Frag PB[2];
#pragma unroll
    for (int h = 0; h < 2; ++h) {
      const unsigned short* pp = &P[wave][l][h * 32 + lg * 16];
      PB[h].q[0] = *(const u32x4*)(pp);
      PB[h].q[1] = *(const u32x4*)(pp + 8);
    }
    // O(c,n) += V(c,m) * P^T(m,n): 16 c-tiles x 2 K-halves
#pragma unroll
    for (int ct = 0; ct < 16; ++ct) {
      const unsigned short* vp = v + ((size_t)b * CC + ct * 16 + l) * HW + m0;
#pragma unroll
      for (int h = 0; h < 2; ++h) {
        Frag A;
        A.q[0] = *(const u32x4*)(vp + h * 32 + lg * 16);
        A.q[1] = *(const u32x4*)(vp + h * 32 + lg * 16 + 8);
        O[ct] = __builtin_amdgcn_wmma_f32_16x16x32_bf16(
            false, A.v, false, PB[h].v, (short)0, O[ct], false, false);
      }
    }
  }

  // ---- split-K merge across the 4 waves ----
  if (lg == 0) { redM[wave][l] = mrun; redL[wave][l] = lrun; }
#pragma unroll
  for (int i = 0; i < CC * 16 / 128; ++i) sO[i * 128 + threadIdx.x] = 0.f;
  __syncthreads();

  float mstar = redM[0][l];
#pragma unroll
  for (int w = 1; w < KSPLIT; ++w) mstar = fmaxf(mstar, redM[w][l]);
  float L = 0.f;
#pragma unroll
  for (int w = 0; w < KSPLIT; ++w) L += redL[w][l] * __expf(redM[w][l] - mstar);
  const float myscale = __expf(mrun - mstar);
  if (threadIdx.x < 16) sInvL[threadIdx.x] = 1.f / L;

#pragma unroll
  for (int ct = 0; ct < 16; ++ct)
#pragma unroll
    for (int j = 0; j < 8; ++j)
      atomicAdd(&sO[(ct * 16 + lg * 8 + j) * 16 + l], O[ct][j] * myscale);
  __syncthreads();

#pragma unroll
  for (int i = 0; i < CC * 16 / 128; ++i) {
    int flat = i * 128 + threadIdx.x;
    int c = flat >> 4, n = flat & 15;
    out[((size_t)b * CC + c) * HW + n0 + n] = sO[flat] * sInvL[n];
  }
}

// ---------------------------------------------------------------------------
extern "C" void kernel_launch(void* const* d_in, const int* in_sizes, int n_in,
                              void* d_out, int out_size, void* d_ws, size_t ws_size,
                              hipStream_t stream) {
  const float* rgb   = (const float*)d_in[0];
  const float* depth = (const float*)d_in[1];
  const float* Wq    = (const float*)d_in[2];
  const float* bq    = (const float*)d_in[3];
  const float* Wk    = (const float*)d_in[4];
  const float* bk    = (const float*)d_in[5];
  const float* Wv    = (const float*)d_in[6];
  const float* bv    = (const float*)d_in[7];
  float* out = (float*)d_out;

  unsigned short* qT = (unsigned short*)d_ws;                 // [B][hw][32] bf16, 1 MB
  unsigned short* kT = qT + (size_t)BB * HW * CQK;            // [B][hw][32] bf16, 1 MB
  unsigned short* vv = kT + (size_t)BB * HW * CQK;            // [B][C][hw]  bf16, 8 MB

  proj_qk_kernel<<<BB * (HW / 64), 128, 0, stream>>>(depth, Wq, bq, Wk, bk, qT, kT);
  proj_v_kernel<<<BB * (HW / 64) * (CC / 64), 128, 0, stream>>>(rgb, Wv, bv, vv);
  attn_kernel<<<BB * (HW / 16), 128, 0, stream>>>(qT, kT, vv, out);
}
